// MultiheadSelfAttention_10565619548380
// MI455X (gfx1250) — matrix operand order repaired and checked by reading, not compiled, not run on hardware
//
#include <hip/hip_runtime.h>

// ---------------------------------------------------------------------------
// MI455X (gfx1250) fused multi-head self-attention, bf16 WMMA + async-LDS DMA.
//   B=2, S=2048, D=1024, H=16, dh=64  (hard-coded per reference)
// Pipeline: cvt(bf16) -> 3x GEMM (Q,K,V) -> RoPE/split -> flash-attn -> GEMM(Wo)
// Staging uses GLOBAL_LOAD_ASYNC_TO_LDS_B128 with double-buffered tiles.
// ---------------------------------------------------------------------------

typedef __attribute__((ext_vector_type(16))) __bf16 v16bf;
typedef __attribute__((ext_vector_type(8)))  float  v8f;

union Frag16 {
  unsigned int   u32[8];
  unsigned short u16[16];
  v16bf          bf;
};

__device__ __forceinline__ unsigned short f2bf(float f) {
  unsigned u = __float_as_uint(f);
  unsigned r = u + 0x7FFFu + ((u >> 16) & 1u);   // round-to-nearest-even
  return (unsigned short)(r >> 16);
}
__device__ __forceinline__ float bf2f(unsigned short h) {
  return __uint_as_float(((unsigned)h) << 16);
}

// LDS offset of a __shared__ object: flat LDS addresses carry the byte offset
// in addr[31:0] (ISA aperture mapping), so truncation yields the DS address.
__device__ __forceinline__ unsigned lds_off(const void* p) {
  return (unsigned)(unsigned long long)p;
}
// Async DMA: 16 bytes per lane, global -> LDS, tracked by ASYNCcnt.
__device__ __forceinline__ void async_ld_b128(unsigned lds_addr, const void* g) {
  asm volatile("global_load_async_to_lds_b128 %0, %1, off"
               :: "v"(lds_addr), "v"(g) : "memory");
}
__device__ __forceinline__ void wait_async0() {
  asm volatile("s_wait_asynccnt 0x0" ::: "memory");
}

// -------------------------------- converts --------------------------------
__global__ void cvt_f32_bf16(const float* __restrict__ in,
                             unsigned short* __restrict__ out, int n) {
  int i = blockIdx.x * blockDim.x + threadIdx.x;
  if (i < n) out[i] = f2bf(in[i]);
}

// ------------------------------ tiled GEMM ---------------------------------
// out[m][n] = sum_k A[m][k] * W[n][k]   (A: MxK bf16, W: NxK bf16)
// Block: 128 threads (4 waves). Block tile 128x64, wave tile 32x64.
// K chunks of 32, double-buffered via async-LDS DMA.
#define TK 32
__global__ __launch_bounds__(128)
void gemm_bf16_wt(const unsigned short* __restrict__ A,
                  const unsigned short* __restrict__ W,
                  float* __restrict__ outF,
                  unsigned short* __restrict__ outB,
                  int M, int N, int K, int bf16out)
{
  __shared__ __align__(16) unsigned short At[2][128 * TK];  // 2 x 8 KB
  __shared__ __align__(16) unsigned short Wt[2][64 * TK];   // 2 x 4 KB

  const int tid  = threadIdx.x;
  const int lane = tid & 31, wave = tid >> 5;
  const int l15  = lane & 15, hi = lane >> 4;
  const int bm   = blockIdx.y * 128, bn = blockIdx.x * 64;
  const int kbA  = hi ? 8 : 0;     // unused: A fragments load at kbB as well
  const int kbB  = hi ? 16 : 0;    // B-fragment K base (contiguous k = 16*hi + s; A uses it too)

  const unsigned At_base = lds_off(&At[0][0]);
  const unsigned Wt_base = lds_off(&Wt[0][0]);

  v8f acc[2][4] = {};

  // stage(buf, kc): A 128x32 (512 b128) + W 64x32 (256 b128)
  auto stage = [&](int buf, int kc) {
    #pragma unroll
    for (int s = 0; s < 4; ++s) {
      int i = tid + s * 128;             // 0..511
      int r = i >> 2, q = i & 3;
      async_ld_b128(At_base + buf * 8192 + i * 16,
                    A + (size_t)(bm + r) * K + kc + q * 8);
    }
    #pragma unroll
    for (int s = 0; s < 2; ++s) {
      int i = tid + s * 128;             // 0..255
      int r = i >> 2, q = i & 3;
      async_ld_b128(Wt_base + buf * 4096 + i * 16,
                    W + (size_t)(bn + r) * K + kc + q * 8);
    }
  };

  stage(0, 0);
  int buf = 0;
  for (int kc = 0; kc < K; kc += TK) {
    wait_async0();          // stage(kc) complete (next stage not yet issued)
    __syncthreads();        // ... and visible to all waves
    if (kc + TK < K) stage(buf ^ 1, kc + TK);   // overlap with compute below

    Frag16 af[2];
    #pragma unroll
    for (int mt = 0; mt < 2; ++mt) {
      const unsigned short* ap = &At[buf][(wave * 32 + mt * 16 + l15) * TK];
      #pragma unroll
      for (int i = 0; i < 8; ++i)
        af[mt].u32[i] = *(const unsigned int*)(ap + kbB + 2 * i);
    }
    #pragma unroll
    for (int nt = 0; nt < 4; ++nt) {
      Frag16 bfr; // B[k][n] = W[bn+nt*16+n][kc+k]
      const unsigned short* wp = &Wt[buf][(nt * 16 + l15) * TK];
      #pragma unroll
      for (int i = 0; i < 8; ++i)
        bfr.u32[i] = *(const unsigned int*)(wp + kbB + 2 * i);
      #pragma unroll
      for (int mt = 0; mt < 2; ++mt)
        acc[mt][nt] = __builtin_amdgcn_wmma_f32_16x16x32_bf16(
            false, af[mt].bf, false, bfr.bf, (short)0, acc[mt][nt], false, false);
    }
    __syncthreads();        // readers done before stage(kc+64) overwrites buf
    buf ^= 1;
  }

  // D layout: lane holds col n=l15, rows r + 8*hi
  #pragma unroll
  for (int mt = 0; mt < 2; ++mt) {
    #pragma unroll
    for (int nt = 0; nt < 4; ++nt) {
      int col = bn + nt * 16 + l15;
      #pragma unroll
      for (int r = 0; r < 8; ++r) {
        int row = bm + wave * 32 + mt * 16 + r + 8 * hi;
        if (bf16out) outB[(size_t)row * N + col] = f2bf(acc[mt][nt][r]);
        else         outF[(size_t)row * N + col] = acc[mt][nt][r];
      }
    }
  }
}

// --------------------------- RoPE + head split -----------------------------
__global__ void rope_split(const unsigned short* __restrict__ Qg,
                           const unsigned short* __restrict__ Kg,
                           const unsigned short* __restrict__ Vg,
                           const int* __restrict__ pos,
                           unsigned short* __restrict__ Qh,
                           unsigned short* __restrict__ Kh,
                           unsigned short* __restrict__ Vh,
                           int Bn, int S, int H, int dh)
{
  int idx = blockIdx.x * blockDim.x + threadIdx.x;
  int half = dh >> 1;
  int total = Bn * S * H * half;
  if (idx >= total) return;
  int c = idx % half;
  int h = (idx / half) % H;
  int s = (idx / (half * H)) % S;
  int b = idx / (half * H * S);
  int D = H * dh;

  size_t src = (size_t)(b * S + s) * D + h * dh + 2 * c;
  size_t dst = ((size_t)(b * H + h) * S + s) * (size_t)dh + 2 * c;

  float p    = (float)pos[b * S + s];
  float freq = __expf(-((float)(2 * c) / (float)dh) * __logf(10000.0f));
  float sn, cs;
  __sincosf(p * freq, &sn, &cs);

  float q1 = bf2f(Qg[src]), q2 = bf2f(Qg[src + 1]);
  Qh[dst]     = f2bf(q1 * cs - q2 * sn);
  Qh[dst + 1] = f2bf(q1 * sn + q2 * cs);
  float k1 = bf2f(Kg[src]), k2 = bf2f(Kg[src + 1]);
  Kh[dst]     = f2bf(k1 * cs - k2 * sn);
  Kh[dst + 1] = f2bf(k1 * sn + k2 * cs);
  Vh[dst]     = Vg[src];
  Vh[dst + 1] = Vg[src + 1];
}

// ------------------------------ flash attention ----------------------------
// Block = 128 threads (4 waves), 64 queries; wave w owns rows w*16..w*16+15.
// Streams 32-key chunks (double-buffered async-LDS), online softmax.
__global__ __launch_bounds__(128)
void flash_attn(const unsigned short* __restrict__ Qh,
                const unsigned short* __restrict__ Kh,
                const unsigned short* __restrict__ Vh,
                unsigned short* __restrict__ Oh,
                int Bn, int H, int S, int dh)
{
  __shared__ __align__(16) unsigned short Qlds[64 * 64];       // 8 KB
  __shared__ __align__(16) unsigned short Klds[2][32 * 64];    // 2 x 4 KB
  __shared__ __align__(16) unsigned short Vlds[2][32 * 64];    // 2 x 4 KB
  __shared__ __align__(16) unsigned short Plds[4][16 * 32];    // 4 KB

  const int tid  = threadIdx.x;
  const int lane = tid & 31, wave = tid >> 5;
  const int l15  = lane & 15, hi = lane >> 4;
  const int kbA  = hi ? 8 : 0;
  const int kbB  = hi ? 16 : 0;
  const int qblk = blockIdx.x * 64;
  const int h    = blockIdx.y, b = blockIdx.z;
  const size_t baseHd = ((size_t)(b * H + h)) * S * dh;

  const unsigned Q_base = lds_off(&Qlds[0]);
  const unsigned K_base = lds_off(&Klds[0][0]);
  const unsigned V_base = lds_off(&Vlds[0][0]);

  auto stageKV = [&](int buf, int kb) {   // 32x64 bf16 each = 256 b128 each
    #pragma unroll
    for (int s = 0; s < 2; ++s) {
      int i = tid + s * 128;
      async_ld_b128(K_base + buf * 4096 + i * 16, Kh + baseHd + (size_t)kb * dh + i * 8);
      async_ld_b128(V_base + buf * 4096 + i * 16, Vh + baseHd + (size_t)kb * dh + i * 8);
    }
  };

  { // stage Q (64x64 = 512 b128) + first K/V chunk
    #pragma unroll
    for (int s = 0; s < 4; ++s) {
      int i = tid + s * 128;
      async_ld_b128(Q_base + i * 16, Qh + baseHd + (size_t)qblk * dh + i * 8);
    }
    stageKV(0, 0);
  }
  wait_async0();
  __syncthreads();

  Frag16 qa[2]; // A-fragments for d-chunks [0,32) and [32,64)
  {
    const unsigned short* qp = Qlds + (wave * 16 + l15) * 64;
    #pragma unroll
    for (int dc = 0; dc < 2; ++dc) {
      #pragma unroll
      for (int i = 0; i < 8; ++i)
        qa[dc].u32[i] = *(const unsigned int*)(qp + dc * 32 + kbB + 2 * i);
    }
  }

  float m_run[8], l_run[8];
  #pragma unroll
  for (int r = 0; r < 8; ++r) { m_run[r] = -1e30f; l_run[r] = 0.f; }
  v8f accO[4] = {};
  const float scale = 0.125f; // 1/sqrt(64)
  const int qrow0 = qblk + wave * 16 + 8 * hi;

  const int nk = qblk + 64;  // causal: keys 0 .. qblk+63
  int buf = 0;
  for (int kb = 0; kb < nk; kb += 32) {
    if (kb) { wait_async0(); __syncthreads(); }     // stage(kb) ready+visible
    if (kb + 32 < nk) stageKV(buf ^ 1, kb + 32);    // overlap next chunk

    // ---- scores: two 16-key tiles, each = 2 WMMAs over dh halves ----
    v8f sc[2] = {};
    #pragma unroll
    for (int t = 0; t < 2; ++t) {
      #pragma unroll
      for (int dc = 0; dc < 2; ++dc) {
        Frag16 kf; // B[d][n] = K[kb + t*16 + n][dc*32 + d]
        const unsigned short* kp = &Klds[buf][(t * 16 + l15) * 64 + dc * 32];
        #pragma unroll
        for (int i = 0; i < 8; ++i)
          kf.u32[i] = *(const unsigned int*)(kp + kbB + 2 * i);
        sc[t] = __builtin_amdgcn_wmma_f32_16x16x32_bf16(
            false, qa[dc].bf, false, kf.bf, (short)0, sc[t], false, false);
      }
    }

    // ---- scale + causal mask ----
    #pragma unroll
    for (int t = 0; t < 2; ++t) {
      int kcol = kb + t * 16 + l15;
      #pragma unroll
      for (int r = 0; r < 8; ++r) {
        float s = sc[t][r] * scale;
        sc[t][r] = (kcol <= qrow0 + r) ? s : -1e30f;
      }
    }

    // ---- online softmax (each row lives in one 16-lane half) ----
    float corr[8];
    #pragma unroll
    for (int r = 0; r < 8; ++r) {
      float v = fmaxf(sc[0][r], sc[1][r]);
      #pragma unroll
      for (int off = 1; off < 16; off <<= 1)
        v = fmaxf(v, __shfl_xor(v, off, 32));
      float mn = fmaxf(m_run[r], v);
      corr[r]  = __expf(m_run[r] - mn);
      m_run[r] = mn;
    }

    unsigned short* pp = &Plds[wave][0];
    float rs[8];
    #pragma unroll
    for (int t = 0; t < 2; ++t) {
      #pragma unroll
      for (int r = 0; r < 8; ++r) {
        float p = __expf(sc[t][r] - m_run[r]);
        if (t == 0) rs[r] = p; else rs[r] += p;
        pp[(r + 8 * hi) * 32 + t * 16 + l15] = f2bf(p);
      }
    }
    #pragma unroll
    for (int r = 0; r < 8; ++r) {
      float v = rs[r];
      #pragma unroll
      for (int off = 1; off < 16; off <<= 1)
        v += __shfl_xor(v, off, 32);
      l_run[r] = corr[r] * l_run[r] + v;
    }
    #pragma unroll
    for (int dt = 0; dt < 4; ++dt)
      #pragma unroll
      for (int r = 0; r < 8; ++r)
        accO[dt][r] *= corr[r];

    // ---- O += P(16x32) x V(32x64) ----
    Frag16 pa;
    {
      const unsigned short* ap = pp + l15 * 32;
      #pragma unroll
      for (int i = 0; i < 8; ++i)
        pa.u32[i] = *(const unsigned int*)(ap + kbB + 2 * i);
    }
    #pragma unroll
    for (int dt = 0; dt < 4; ++dt) {
      Frag16 vf; // B[k][n] = V[kb+k][dt*16+n]
      #pragma unroll
      for (int e = 0; e < 16; ++e)
        vf.u16[e] = Vlds[buf][(kbB + e) * 64 + dt * 16 + l15];
      accO[dt] = __builtin_amdgcn_wmma_f32_16x16x32_bf16(
          false, pa.bf, false, vf.bf, (short)0, accO[dt], false, false);
    }

    __syncthreads();   // readers done before stage(kb+64) overwrites this buf
    buf ^= 1;
  }

  // ---- epilogue: divide by row sum, write [B][S][H][dh] bf16 ----
  const int D = H * dh;
  #pragma unroll
  for (int r = 0; r < 8; ++r) {
    float inv = 1.0f / l_run[r];
    int row = b * S + (qrow0 + r);
    #pragma unroll
    for (int dt = 0; dt < 4; ++dt) {
      int col = h * dh + dt * 16 + l15;
      Oh[(size_t)row * D + col] = f2bf(accO[dt][r] * inv);
    }
  }
}

// ------------------------------- launcher ----------------------------------
extern "C" void kernel_launch(void* const* d_in, const int* in_sizes, int n_in,
                              void* d_out, int out_size, void* d_ws, size_t ws_size,
                              hipStream_t stream) {
  (void)in_sizes; (void)n_in; (void)out_size; (void)ws_size;
  const int B = 2, S = 2048, D = 1024, H = 16, dh = 64;
  const int M = B * S;

  const float* x  = (const float*)d_in[0];
  const float* Wq = (const float*)d_in[1];
  const float* Wk = (const float*)d_in[2];
  const float* Wv = (const float*)d_in[3];
  const float* Wo = (const float*)d_in[4];
  const int* tpos = (const int*)d_in[5];
  float* out = (float*)d_out;

  // workspace carve-up (~72 MB total)
  char* ws = (char*)d_ws;
  size_t off = 0;
  auto carve = [&](size_t bytes) { char* p = ws + off; off += (bytes + 255) & ~(size_t)255; return p; };
  unsigned short* xbf = (unsigned short*)carve((size_t)M * D * 2);
  unsigned short* WqB = (unsigned short*)carve((size_t)D * D * 2);
  unsigned short* WkB = (unsigned short*)carve((size_t)D * D * 2);
  unsigned short* WvB = (unsigned short*)carve((size_t)D * D * 2);
  unsigned short* WoB = (unsigned short*)carve((size_t)D * D * 2);
  unsigned short* Qg  = (unsigned short*)carve((size_t)M * D * 2);
  unsigned short* Kg  = (unsigned short*)carve((size_t)M * D * 2);
  unsigned short* Vg  = (unsigned short*)carve((size_t)M * D * 2);
  unsigned short* QhB = (unsigned short*)carve((size_t)M * D * 2);
  unsigned short* KhB = (unsigned short*)carve((size_t)M * D * 2);
  unsigned short* VhB = (unsigned short*)carve((size_t)M * D * 2);
  unsigned short* OhB = (unsigned short*)carve((size_t)M * D * 2);

  // 1) fp32 -> bf16
  {
    int n = M * D;
    cvt_f32_bf16<<<(n + 255) / 256, 256, 0, stream>>>(x, xbf, n);
    int nw = D * D;
    cvt_f32_bf16<<<(nw + 255) / 256, 256, 0, stream>>>(Wq, WqB, nw);
    cvt_f32_bf16<<<(nw + 255) / 256, 256, 0, stream>>>(Wk, WkB, nw);
    cvt_f32_bf16<<<(nw + 255) / 256, 256, 0, stream>>>(Wv, WvB, nw);
    cvt_f32_bf16<<<(nw + 255) / 256, 256, 0, stream>>>(Wo, WoB, nw);
  }

  // 2) Q/K/V projections (bf16 out); block tile 128x64
  dim3 gg(D / 64, M / 128);
  gemm_bf16_wt<<<gg, 128, 0, stream>>>(xbf, WqB, nullptr, Qg, M, D, D, 1);
  gemm_bf16_wt<<<gg, 128, 0, stream>>>(xbf, WkB, nullptr, Kg, M, D, D, 1);
  gemm_bf16_wt<<<gg, 128, 0, stream>>>(xbf, WvB, nullptr, Vg, M, D, D, 1);

  // 3) RoPE + head split
  {
    int total = B * S * H * (dh / 2);
    rope_split<<<(total + 255) / 256, 256, 0, stream>>>(
        Qg, Kg, Vg, tpos, QhB, KhB, VhB, B, S, H, dh);
  }

  // 4) causal flash attention
  {
    dim3 grid(S / 64, H, B);
    flash_attn<<<grid, 128, 0, stream>>>(QhB, KhB, VhB, OhB, B, H, S, dh);
  }

  // 5) output projection (fp32 out -> d_out)
  gemm_bf16_wt<<<gg, 128, 0, stream>>>(OhB, WoB, out, nullptr, M, D, D, 0);
}
